// MultilayerLSTM_75385265979394
// MI455X (gfx1250) — matrix operand order repaired and checked by reading, not compiled, not run on hardware
//
#include <hip/hip_runtime.h>

typedef _Float16 half8  __attribute__((ext_vector_type(8)));
typedef _Float16 v16h   __attribute__((ext_vector_type(16)));
typedef float    v8f    __attribute__((ext_vector_type(8)));

#define NWG 32          // persistent grid: 32 WGs x 256 threads (8 waves each)
#define DH  1024
#define BSZ 16
#define SEQ 512
#define DIN 512

// ---------------------------------------------------------------------------
// helpers
// ---------------------------------------------------------------------------
__device__ __forceinline__ v16h cat16(half8 lo, half8 hi) {
  return __builtin_shufflevector(lo, hi, 0,1,2,3,4,5,6,7,8,9,10,11,12,13,14,15);
}

__device__ __forceinline__ float sigf(float x) { return 1.0f / (1.0f + __expf(-x)); }

__device__ __forceinline__ v8f wmma_f16(v16h a, v16h b, v8f c) {
  return __builtin_amdgcn_wmma_f32_16x16x32_f16(false, a, false, b, (short)0, c,
                                                false, false);
}

// load a 16-half A/B fragment from base + 32-bit element offset (saddr+voffset form)
__device__ __forceinline__ v16h frag16(const _Float16* __restrict__ base,
                                       unsigned off0, unsigned off1) {
  return cat16(*(const half8*)(base + off0), *(const half8*)(base + off1));
}

// grid-wide barrier: generation-counter scheme, thread0 spins with s_sleep
__device__ __forceinline__ void grid_barrier(int* arrive, int* release) {
  __threadfence();           // make this thread's global stores visible (agent)
  __syncthreads();
  if (threadIdx.x == 0) {
    int ticket = __hip_atomic_fetch_add(arrive, 1, __ATOMIC_RELAXED, __HIP_MEMORY_SCOPE_AGENT);
    int gen = ticket / NWG + 1;
    if ((ticket % NWG) == NWG - 1)
      __hip_atomic_fetch_add(release, 1, __ATOMIC_RELEASE, __HIP_MEMORY_SCOPE_AGENT);
    while (__hip_atomic_load(release, __ATOMIC_ACQUIRE, __HIP_MEMORY_SCOPE_AGENT) < gen)
      __builtin_amdgcn_s_sleep(1);
  }
  __syncthreads();
}

// ---------------------------------------------------------------------------
// prep kernels
// ---------------------------------------------------------------------------

// x_h[m][t][k] = fp16(emb[tokens[m,t]][k])
__global__ void embed_kernel(const int* __restrict__ tokens,
                             const float* __restrict__ emb,
                             _Float16* __restrict__ xh) {
  size_t i = (size_t)blockIdx.x * 256 + threadIdx.x;      // over 16*512*512
  if (i >= (size_t)BSZ * SEQ * DIN) return;
  int k = (int)(i & (DIN - 1));
  size_t bt = i >> 9;                                     // m*512 + t
  int tok = tokens[bt];
  xh[i] = (_Float16)emb[(size_t)tok * DIN + k];
}

// pack W[(K),1024] fp32 (row=k, col=h) into per-lane WMMA B-fragment order:
// block (ht, gate, kb): 512 halves; offset = lane*16 + j holds W[kb*32 + klane + (j&7) + 16*(j>>3)][ht*16 + (lane&15)]
// klane = 0 for lanes 0-15, 8 for lanes 16-31  (ISA 16-bit B layout, K=32)
__global__ void pack_w_kernel(const float* __restrict__ W,
                              _Float16* __restrict__ dst,
                              int KB, int gate) {
  size_t i = (size_t)blockIdx.x * 256 + threadIdx.x;
  size_t total = (size_t)64 * KB * 512;
  if (i >= total) return;
  int j    = (int)(i & 15);
  int lane = (int)((i >> 4) & 31);
  size_t rem = i >> 9;
  int kb = (int)(rem % KB);
  int ht = (int)(rem / KB);
  int k = kb * 32 + ((lane >> 4) << 3) + (j & 7) + ((j >> 3) << 4);
  int n = ht * 16 + (lane & 15);
  dst[((size_t)(ht * 4 + gate) * KB + kb) * 512 + lane * 16 + j] =
      (_Float16)W[(size_t)k * DH + n];
}

__global__ void pack_bias_kernel(const float* b0i, const float* b0f, const float* b0g, const float* b0o,
                                 const float* b1i, const float* b1f, const float* b1g, const float* b1o,
                                 float* __restrict__ bcomb) {
  int i = blockIdx.x * 256 + threadIdx.x;                 // 2*4*1024 = 8192
  if (i >= 8192) return;
  const float* src[8] = {b0i, b0f, b0g, b0o, b1i, b1f, b1g, b1o};
  bcomb[i] = src[i >> 10][i & 1023];
}

// ---------------------------------------------------------------------------
// persistent LSTM recurrence kernel
// ---------------------------------------------------------------------------
__global__ __launch_bounds__(256) void lstm_seq_kernel(
    const _Float16* __restrict__ xh,      // [16][512][512]
    const _Float16* __restrict__ w0p,     // packed [64*4*48*512]
    const _Float16* __restrict__ w1p,     // packed [64*4*64*512]
    const float*    __restrict__ bcomb,   // [2][4][1024]
    _Float16* __restrict__ h0buf,         // ping-pong [2][16][1024]
    _Float16* __restrict__ h1buf,         // ping-pong [2][16][1024]
    _Float16* __restrict__ h1arc,         // [512][16][1024]
    int* __restrict__ ctr)
{
  __shared__ float gates[2][4][16][16];   // [h-tile][gate][m][n]
  __shared__ float cst[2][2][16][16];     // [layer][h-tile][m][n]

  const int tid  = threadIdx.x;
  const int wg   = blockIdx.x;
  const int lane = tid & 31;
  const int wave = tid >> 5;
  const int tile = wave >> 2;             // 0..1
  const int gate = wave & 3;              // 0..3 (i,f,g,o)
  const int ht   = wg * 2 + tile;         // 0..63 hidden tile
  const int mrow = lane & 15;             // A row / C column
  const unsigned koff = (unsigned)((lane >> 4) * 8); // A chunk base within K-block of 32
  const int mb   = (lane >> 4) * 8;       // C/D fragment row base

  for (int i = tid; i < 2 * 2 * 256; i += 256) ((float*)cst)[i] = 0.0f;
  __syncthreads();

  int* arrive  = ctr;
  int* release = ctr + 16;

  const float bv0 = bcomb[gate * DH + ht * 16 + mrow];
  const float bv1 = bcomb[4096 + gate * DH + ht * 16 + mrow];
  // 32-bit element offsets into the packed weight arrays (uniform base -> saddr form)
  const unsigned w0off = (unsigned)(ht * 4 + gate) * 48u * 512u + (unsigned)lane * 16u;
  const unsigned w1off = (unsigned)(ht * 4 + gate) * 64u * 512u + (unsigned)lane * 16u;
  const unsigned xrow_base = (unsigned)(mrow * SEQ) * DIN + koff;   // + t*DIN at use
  const unsigned hrow      = (unsigned)mrow * DH + koff;            // within one H image

  for (int t = 0; t < SEQ; ++t) {
    const unsigned cur = (unsigned)(t & 1), prv = cur ^ 1u;
    const unsigned h0p_off = prv * (BSZ * DH) + hrow;   // H0_prev  (element offsets)
    const unsigned h0c_off = cur * (BSZ * DH) + hrow;   // H0_cur
    const unsigned h1p_off = prv * (BSZ * DH) + hrow;   // H1_prev
    _Float16* h0c = h0buf + (size_t)cur * BSZ * DH;
    _Float16* h1c = h1buf + (size_t)cur * BSZ * DH;

    // ================= layer 0 : K = 1536 (48 blocks of 32) =================
    {
      v8f acc0 = {}, acc1 = {};
      const unsigned xrow = xrow_base + (unsigned)t * DIN;
      // blocks 0..15: x part (k = 0..511)
      #pragma unroll 4
      for (unsigned kb = 0; kb < 16; kb += 2) {
        acc0 = wmma_f16(frag16(xh, xrow + kb * 32u, xrow + kb * 32u + 16u),
                        frag16(w0p, w0off + kb * 512u, w0off + kb * 512u + 8u), acc0);
        acc1 = wmma_f16(frag16(xh, xrow + kb * 32u + 32u, xrow + kb * 32u + 48u),
                        frag16(w0p, w0off + kb * 512u + 512u, w0off + kb * 512u + 520u), acc1);
      }
      // blocks 16..47: recurrent part (k = 512..1535 -> H0_prev[0..1023])
      #pragma unroll 4
      for (unsigned kb = 16; kb < 48; kb += 2) {
        const unsigned hk = h0p_off + (kb - 16u) * 32u;
        acc0 = wmma_f16(frag16(h0buf, hk, hk + 16u),
                        frag16(w0p, w0off + kb * 512u, w0off + kb * 512u + 8u), acc0);
        acc1 = wmma_f16(frag16(h0buf, hk + 32u, hk + 48u),
                        frag16(w0p, w0off + kb * 512u + 512u, w0off + kb * 512u + 520u), acc1);
      }
      v8f acc = acc0 + acc1;
      #pragma unroll
      for (int r = 0; r < 8; ++r) gates[tile][gate][mb + r][mrow] = acc[r] + bv0;
    }
    __syncthreads();
    // fuse gates -> C0, H0
    #pragma unroll
    for (int s = 0; s < 2; ++s) {
      int idx = tid + s * 256;                       // over [2][16][16]
      int tl = idx >> 8, mm = (idx >> 4) & 15, nn = idx & 15;
      float iv = sigf(gates[tl][0][mm][nn]);
      float fv = sigf(gates[tl][1][mm][nn]);
      float gv = tanhf(gates[tl][2][mm][nn]);
      float ov = sigf(gates[tl][3][mm][nn]);
      float c = fv * cst[0][tl][mm][nn] + iv * gv;
      cst[0][tl][mm][nn] = c;
      h0c[mm * DH + (wg * 2 + tl) * 16 + nn] = (_Float16)(ov * tanhf(c));
    }
    grid_barrier(arrive, release);

    // ================= layer 1 : K = 2048 (64 blocks of 32) =================
    {
      v8f acc0 = {}, acc1 = {};
      // blocks 0..31: H0_cur part (k = 0..1023)
      #pragma unroll 4
      for (unsigned kb = 0; kb < 32; kb += 2) {
        const unsigned hk = h0c_off + kb * 32u;
        acc0 = wmma_f16(frag16(h0buf, hk, hk + 16u),
                        frag16(w1p, w1off + kb * 512u, w1off + kb * 512u + 8u), acc0);
        acc1 = wmma_f16(frag16(h0buf, hk + 32u, hk + 48u),
                        frag16(w1p, w1off + kb * 512u + 512u, w1off + kb * 512u + 520u), acc1);
      }
      // blocks 32..63: H1_prev part (k = 1024..2047)
      #pragma unroll 4
      for (unsigned kb = 32; kb < 64; kb += 2) {
        const unsigned hk = h1p_off + (kb - 32u) * 32u;
        acc0 = wmma_f16(frag16(h1buf, hk, hk + 16u),
                        frag16(w1p, w1off + kb * 512u, w1off + kb * 512u + 8u), acc0);
        acc1 = wmma_f16(frag16(h1buf, hk + 32u, hk + 48u),
                        frag16(w1p, w1off + kb * 512u + 512u, w1off + kb * 512u + 520u), acc1);
      }
      v8f acc = acc0 + acc1;
      #pragma unroll
      for (int r = 0; r < 8; ++r) gates[tile][gate][mb + r][mrow] = acc[r] + bv1;
    }
    __syncthreads();
    // fuse gates -> C1, H1 (+ archive for FC head)
    #pragma unroll
    for (int s = 0; s < 2; ++s) {
      int idx = tid + s * 256;
      int tl = idx >> 8, mm = (idx >> 4) & 15, nn = idx & 15;
      float iv = sigf(gates[tl][0][mm][nn]);
      float fv = sigf(gates[tl][1][mm][nn]);
      float gv = tanhf(gates[tl][2][mm][nn]);
      float ov = sigf(gates[tl][3][mm][nn]);
      float c = fv * cst[1][tl][mm][nn] + iv * gv;
      cst[1][tl][mm][nn] = c;
      _Float16 h = (_Float16)(ov * tanhf(c));
      int hcol = (wg * 2 + tl) * 16 + nn;
      h1c[mm * DH + hcol] = h;
      h1arc[((size_t)t * BSZ + mm) * DH + hcol] = h;
    }
    grid_barrier(arrive, release);
  }
}

// ---------------------------------------------------------------------------
// FC head + 2-class log_softmax (fp32)
// ---------------------------------------------------------------------------
__global__ void head_kernel(const _Float16* __restrict__ h1arc,
                            const float* __restrict__ Wfc,
                            const float* __restrict__ bfc,
                            float* __restrict__ out) {
  int i = blockIdx.x * 256 + threadIdx.x;     // i = b*512 + t, 8192 rows
  if (i >= BSZ * SEQ) return;
  int b = i >> 9, t = i & (SEQ - 1);
  const _Float16* hrow = h1arc + ((size_t)t * BSZ + b) * DH;
  float l0 = bfc[0], l1 = bfc[1];
  for (int h = 0; h < DH; h += 8) {
    half8 hv = *(const half8*)(hrow + h);
    #pragma unroll
    for (int j = 0; j < 8; ++j) {
      float x = (float)hv[j];
      l0 += x * Wfc[(h + j) * 2 + 0];
      l1 += x * Wfc[(h + j) * 2 + 1];
    }
  }
  float mx = fmaxf(l0, l1);
  float lse = mx + logf(expf(l0 - mx) + expf(l1 - mx));
  out[(size_t)i * 2 + 0] = l0 - lse;
  out[(size_t)i * 2 + 1] = l1 - lse;
}

// ---------------------------------------------------------------------------
// launch
// ---------------------------------------------------------------------------
extern "C" void kernel_launch(void* const* d_in, const int* in_sizes, int n_in,
                              void* d_out, int out_size, void* d_ws, size_t ws_size,
                              hipStream_t stream) {
  (void)in_sizes; (void)n_in; (void)out_size; (void)ws_size;

  const int*   tokens = (const int*)d_in[0];
  const float* emb    = (const float*)d_in[1];
  const float* W0[4]  = {(const float*)d_in[2], (const float*)d_in[4],
                         (const float*)d_in[6], (const float*)d_in[8]};
  const float* b0[4]  = {(const float*)d_in[3], (const float*)d_in[5],
                         (const float*)d_in[7], (const float*)d_in[9]};
  const float* W1[4]  = {(const float*)d_in[10], (const float*)d_in[12],
                         (const float*)d_in[14], (const float*)d_in[16]};
  const float* b1[4]  = {(const float*)d_in[11], (const float*)d_in[13],
                         (const float*)d_in[15], (const float*)d_in[17]};
  const float* Wfc    = (const float*)d_in[18];
  const float* bfc    = (const float*)d_in[19];

  char* ws = (char*)d_ws;
  size_t off = 0;
  int*      ctr   = (int*)(ws + off);       off += 256;
  _Float16* xh    = (_Float16*)(ws + off);  off += (size_t)BSZ * SEQ * DIN * 2;   // 8 MB
  _Float16* w0p   = (_Float16*)(ws + off);  off += (size_t)64 * 4 * 48 * 512 * 2; // 12 MB
  _Float16* w1p   = (_Float16*)(ws + off);  off += (size_t)64 * 4 * 64 * 512 * 2; // 16 MB
  float*    bcomb = (float*)(ws + off);     off += 8192 * 4;
  _Float16* h0buf = (_Float16*)(ws + off);  off += (size_t)2 * BSZ * DH * 2;
  _Float16* h1buf = (_Float16*)(ws + off);  off += (size_t)2 * BSZ * DH * 2;
  _Float16* h1arc = (_Float16*)(ws + off);  off += (size_t)SEQ * BSZ * DH * 2;    // 16 MB

  hipMemsetAsync(ctr, 0, 256, stream);
  hipMemsetAsync(h0buf, 0, (size_t)2 * BSZ * DH * 2, stream);
  hipMemsetAsync(h1buf, 0, (size_t)2 * BSZ * DH * 2, stream);

  embed_kernel<<<(BSZ * SEQ * DIN) / 256, 256, 0, stream>>>(tokens, emb, xh);
  pack_bias_kernel<<<32, 256, 0, stream>>>(b0[0], b0[1], b0[2], b0[3],
                                           b1[0], b1[1], b1[2], b1[3], bcomb);
  for (int g = 0; g < 4; ++g)
    pack_w_kernel<<<(64 * 48 * 512) / 256, 256, 0, stream>>>(W0[g], w0p, 48, g);
  for (int g = 0; g < 4; ++g)
    pack_w_kernel<<<(64 * 64 * 512) / 256, 256, 0, stream>>>(W1[g], w1p, 64, g);

  lstm_seq_kernel<<<NWG, 256, 0, stream>>>(xh, w0p, w1p, bcomb,
                                           h0buf, h1buf, h1arc, ctr);

  head_kernel<<<(BSZ * SEQ) / 256, 256, 0, stream>>>(h1arc, Wfc, bfc, (float*)d_out);
}
